// RecurrentGCN_28424093565280
// MI455X (gfx1250) — matrix operand order repaired and checked by reading, not compiled, not run on hardware
//
#include <hip/hip_runtime.h>
#include <math.h>

typedef __attribute__((ext_vector_type(16))) _Float16 v16h;
typedef __attribute__((ext_vector_type(8)))  _Float16 v8h;
typedef __attribute__((ext_vector_type(2)))  _Float16 v2h;
typedef __attribute__((ext_vector_type(8)))  float    v8f;
typedef __attribute__((ext_vector_type(4)))  float    v4f;

#define EMBED 64   // NODE_DIM == EMBED == 64

// ---------------------------------------------------------------------------
// WMMA helper (CDNA5 16x16x32 f16 -> f32)
// ---------------------------------------------------------------------------
__device__ __forceinline__ v8f wmma_f16(v16h a, v16h b, v8f c) {
  return __builtin_amdgcn_wmma_f32_16x16x32_f16(
      /*neg_a=*/false, a, /*neg_b=*/false, b,
      /*c_mod=*/(short)0, c, /*reuse_a=*/false, /*reuse_b=*/false);
}

// A fragment: 16x32 f16 from row-major f32. Lane L and L+16 carry row M=L%16;
// lane's 16 halfs are two runs of 8 consecutive K -> 4x float4 loads.
// Clamped variant for exact-size input arrays (x, h).
__device__ __forceinline__ v16h load_a_clamp(const float* __restrict__ A,
                                             int m0, int kc, int lane, int num) {
  int r = lane & 15, hh = lane >> 4;
  int row = m0 + r; if (row >= num) row = num - 1;
  const v4f* p4 = (const v4f*)(A + (size_t)row * EMBED);
  int q0 = (kc + hh * 8) >> 2;
  int q1 = (kc + 16 + hh * 8) >> 2;
  v4f x0 = p4[q0], x1 = p4[q0 + 1], x2 = p4[q1], x3 = p4[q1 + 1];
  v16h f;
#pragma unroll
  for (int i = 0; i < 4; ++i) {
    f[i]      = (_Float16)x0[i];
    f[4 + i]  = (_Float16)x1[i];
    f[8 + i]  = (_Float16)x2[i];
    f[12 + i] = (_Float16)x3[i];
  }
  return f;
}

// Unclamped variant for workspace arrays padded to num_pad rows.
__device__ __forceinline__ v16h load_a_ws(const float* __restrict__ A,
                                          int m0, int kc, int lane) {
  int r = lane & 15, hh = lane >> 4;
  const v4f* p4 = (const v4f*)(A + (size_t)(m0 + r) * EMBED);
  int q0 = (kc + hh * 8) >> 2;
  int q1 = (kc + 16 + hh * 8) >> 2;
  v4f x0 = p4[q0], x1 = p4[q0 + 1], x2 = p4[q1], x3 = p4[q1 + 1];
  v16h f;
#pragma unroll
  for (int i = 0; i < 4; ++i) {
    f[i]      = (_Float16)x0[i];
    f[4 + i]  = (_Float16)x1[i];
    f[8 + i]  = (_Float16)x2[i];
    f[12 + i] = (_Float16)x3[i];
  }
  return f;
}

// A fragment = elementwise H[row,k] * R[row,k] (f32 input * f16 workspace).
__device__ __forceinline__ v16h load_a_hr(const float* __restrict__ H,
                                          const _Float16* __restrict__ R,
                                          int m0, int kc, int lane, int num) {
  int r = lane & 15, hh = lane >> 4;
  int row = m0 + r; if (row >= num) row = num - 1;
  const v4f* p4 = (const v4f*)(H + (size_t)row * EMBED);
  const v8h* r8 = (const v8h*)(R + (size_t)row * EMBED);
  int k0 = kc + hh * 8, k1 = kc + 16 + hh * 8;
  v4f x0 = p4[k0 >> 2], x1 = p4[(k0 >> 2) + 1];
  v4f x2 = p4[k1 >> 2], x3 = p4[(k1 >> 2) + 1];
  v8h ra = r8[k0 >> 3], rb = r8[k1 >> 3];
  v16h f;
#pragma unroll
  for (int i = 0; i < 4; ++i) {
    f[i]      = (_Float16)(x0[i] * (float)ra[i]);
    f[4 + i]  = (_Float16)(x1[i] * (float)ra[4 + i]);
    f[8 + i]  = (_Float16)(x2[i] * (float)rb[i]);
    f[12 + i] = (_Float16)(x3[i] * (float)rb[4 + i]);
  }
  return f;
}

// Stage weight W[K,64] (row-major f32) into LDS as f16, transposed n-major:
// lds[n*K + k] = W[k*64 + n]. Global reads coalesced (linear over W), the
// transpose happens on the LDS store side. B fragment = two runs of 8 halfs.
template <int K>
__device__ __forceinline__ void stage_w(const float* __restrict__ W,
                                        _Float16* __restrict__ lds) {
  for (int i = threadIdx.x; i < K * 64; i += blockDim.x) {
    int k = i >> 6, n = i & 63;
    lds[(size_t)n * K + k] = (_Float16)W[i];
  }
}

template <int K>
__device__ __forceinline__ v16h load_b_lds(const _Float16* __restrict__ lds,
                                           int kc, int nc, int lane) {
  int r = lane & 15, hh = lane >> 4;
  const _Float16* p = lds + (size_t)(nc + r) * K + kc + hh * 8;
  v8h lo = *(const v8h*)p, hi = *(const v8h*)(p + 16);
  return __builtin_shufflevector(lo, hi, 0,1,2,3,4,5,6,7,8,9,10,11,12,13,14,15);
}

// C fragment element j -> (row = m0 + (lane>>4)*8 + j, col = nc + lane%16)

// ---------------------------------------------------------------------------
__global__ void k_deg_init(float* __restrict__ deg, int num_pad) {
  int i = blockIdx.x * blockDim.x + threadIdx.x;
  if (i < num_pad) deg[i] = 1.0f;
}

__global__ void k_deg_accum(float* __restrict__ deg, const int* __restrict__ dst,
                            const float* __restrict__ ew, int E) {
  int e = blockIdx.x * blockDim.x + threadIdx.x;
  if (e < E) atomicAdd(&deg[dst[e]], ew[e]);
}

__global__ void k_dinv(float* __restrict__ deg, int num_pad) {
  int i = blockIdx.x * blockDim.x + threadIdx.x;
  if (i < num_pad) {
    float d = deg[i];
    deg[i] = (d > 0.0f) ? rsqrtf(d) : 0.0f;
  }
}

// ---------------------------------------------------------------------------
// K4: three GCN projections via WMMA. Writes xw (f16, for edge gathers) and
//     agg = xw*dinv^2 + b. All stores unconditional (workspace padded).
// ---------------------------------------------------------------------------
__global__ void __launch_bounds__(128)
k_conv_gemm(const float* __restrict__ x, const float* __restrict__ dinv,
            const float* __restrict__ Wz, const float* __restrict__ bz,
            const float* __restrict__ Wr, const float* __restrict__ br,
            const float* __restrict__ Wh, const float* __restrict__ bh,
            _Float16* __restrict__ xwz, _Float16* __restrict__ xwr,
            _Float16* __restrict__ xwh,
            float* __restrict__ aggz, float* __restrict__ aggr,
            float* __restrict__ aggh, int num, int num_pad) {
  __shared__ _Float16 sw[3][64 * 64];
  stage_w<64>(Wz, sw[0]);
  stage_w<64>(Wr, sw[1]);
  stage_w<64>(Wh, sw[2]);
  __syncthreads();

  int wave = (blockIdx.x * blockDim.x + threadIdx.x) >> 5;
  int lane = threadIdx.x & 31;
  int m0 = wave * 16;
  if (m0 >= num_pad) return;
  int r = lane & 15, hh = lane >> 4;

  v16h a0 = load_a_clamp(x, m0, 0, lane, num);
  v16h a1 = load_a_clamp(x, m0, 32, lane, num);

  // 8 contiguous dinv values per lane (padded array -> no clamp, 2x b128)
  const v4f* dv = (const v4f*)(dinv + m0 + hh * 8);
  v4f d0 = dv[0], d1 = dv[1];
  float di[8];
#pragma unroll
  for (int i = 0; i < 4; ++i) { di[i] = d0[i]; di[4 + i] = d1[i]; }

  const float* bs[3]   = {bz, br, bh};
  _Float16*    xws[3]  = {xwz, xwr, xwh};
  float*       aggs[3] = {aggz, aggr, aggh};

#pragma unroll
  for (int mat = 0; mat < 3; ++mat) {
#pragma unroll
    for (int nt = 0; nt < 4; ++nt) {
      int nc = nt * 16;
      int col = nc + r;
      float bias = bs[mat][col];
      v8f c = {};
      c = wmma_f16(a0, load_b_lds<64>(sw[mat], 0, nc, lane), c);
      c = wmma_f16(a1, load_b_lds<64>(sw[mat], 32, nc, lane), c);
#pragma unroll
      for (int j = 0; j < 8; ++j) {
        size_t idx = (size_t)(m0 + hh * 8 + j) * EMBED + col;
        float v = c[j];
        xws[mat][idx]  = (_Float16)v;
        aggs[mat][idx] = v * di[j] * di[j] + bias;
      }
    }
  }
}

// ---------------------------------------------------------------------------
// K5: fused edge scatter. One wave per edge, lane owns dims {2l, 2l+1}:
//     3x b32 gather + 6x global_atomic_add_f32.
// ---------------------------------------------------------------------------
__global__ void __launch_bounds__(256)
k_edge_scatter(const int* __restrict__ src, const int* __restrict__ dst,
               const float* __restrict__ ew, const float* __restrict__ dinv,
               const _Float16* __restrict__ xwz, const _Float16* __restrict__ xwr,
               const _Float16* __restrict__ xwh,
               float* __restrict__ aggz, float* __restrict__ aggr,
               float* __restrict__ aggh, int E) {
  int e = (blockIdx.x * blockDim.x + threadIdx.x) >> 5;
  int lane = threadIdx.x & 31;
  if (e >= E) return;
  int s = src[e];
  int d = dst[e];
  float n = dinv[s] * ew[e] * dinv[d];
  size_t sb = (size_t)s * EMBED;
  size_t db = (size_t)d * EMBED;
  int k = lane * 2;
  v2h z2 = *(const v2h*)(xwz + sb + k);
  v2h r2 = *(const v2h*)(xwr + sb + k);
  v2h h2 = *(const v2h*)(xwh + sb + k);
  atomicAdd(&aggz[db + k],     (float)z2[0] * n);
  atomicAdd(&aggz[db + k + 1], (float)z2[1] * n);
  atomicAdd(&aggr[db + k],     (float)r2[0] * n);
  atomicAdd(&aggr[db + k + 1], (float)r2[1] * n);
  atomicAdd(&aggh[db + k],     (float)h2[0] * n);
  atomicAdd(&aggh[db + k + 1], (float)h2[1] * n);
}

// ---------------------------------------------------------------------------
// K6: Z = sigmoid([aggz|h] @ W_lz + b_lz), R = sigmoid([aggr|h] @ W_lr + b_lr)
//     Z/R land in padded workspace -> unconditional stores.
// ---------------------------------------------------------------------------
__global__ void __launch_bounds__(128)
k_gate_zr(const float* __restrict__ aggz, const float* __restrict__ aggr,
          const float* __restrict__ h,
          const float* __restrict__ Wlz, const float* __restrict__ blz,
          const float* __restrict__ Wlr, const float* __restrict__ blr,
          _Float16* __restrict__ Zbuf, _Float16* __restrict__ Rbuf,
          int num, int num_pad) {
  __shared__ _Float16 swz[128 * 64];
  __shared__ _Float16 swr[128 * 64];
  stage_w<128>(Wlz, swz);
  stage_w<128>(Wlr, swr);
  __syncthreads();

  int wave = (blockIdx.x * blockDim.x + threadIdx.x) >> 5;
  int lane = threadIdx.x & 31;
  int m0 = wave * 16;
  if (m0 >= num_pad) return;
  int r = lane & 15, hh = lane >> 4;

  v16h az0 = load_a_ws(aggz, m0, 0, lane);
  v16h az1 = load_a_ws(aggz, m0, 32, lane);
  v16h ar0 = load_a_ws(aggr, m0, 0, lane);
  v16h ar1 = load_a_ws(aggr, m0, 32, lane);
  v16h ah0 = load_a_clamp(h, m0, 0, lane, num);
  v16h ah1 = load_a_clamp(h, m0, 32, lane, num);

#pragma unroll
  for (int nt = 0; nt < 4; ++nt) {
    int nc = nt * 16;
    int col = nc + r;
    v8f cz = {};
    cz = wmma_f16(az0, load_b_lds<128>(swz, 0,  nc, lane), cz);
    cz = wmma_f16(az1, load_b_lds<128>(swz, 32, nc, lane), cz);
    cz = wmma_f16(ah0, load_b_lds<128>(swz, 64, nc, lane), cz);
    cz = wmma_f16(ah1, load_b_lds<128>(swz, 96, nc, lane), cz);
    v8f cr = {};
    cr = wmma_f16(ar0, load_b_lds<128>(swr, 0,  nc, lane), cr);
    cr = wmma_f16(ar1, load_b_lds<128>(swr, 32, nc, lane), cr);
    cr = wmma_f16(ah0, load_b_lds<128>(swr, 64, nc, lane), cr);
    cr = wmma_f16(ah1, load_b_lds<128>(swr, 96, nc, lane), cr);
    float bz_ = blz[col], br_ = blr[col];
#pragma unroll
    for (int j = 0; j < 8; ++j) {
      size_t idx = (size_t)(m0 + hh * 8 + j) * EMBED + col;
      Zbuf[idx] = (_Float16)(1.0f / (1.0f + __expf(-(cz[j] + bz_))));
      Rbuf[idx] = (_Float16)(1.0f / (1.0f + __expf(-(cr[j] + br_))));
    }
  }
}

// ---------------------------------------------------------------------------
// K7: H~ = tanh([aggh | h*R] @ W_lh + b_lh);  h0 = Z*h + (1-Z)*H~
//     Writes real output -> wave-uniform full/tail split.
// ---------------------------------------------------------------------------
__global__ void __launch_bounds__(128)
k_gate_h(const float* __restrict__ aggh, const float* __restrict__ h,
         const _Float16* __restrict__ Zbuf, const _Float16* __restrict__ Rbuf,
         const float* __restrict__ Wlh, const float* __restrict__ blh,
         float* __restrict__ h0out, int num, int num_pad) {
  __shared__ _Float16 swh[128 * 64];
  stage_w<128>(Wlh, swh);
  __syncthreads();

  int wave = (blockIdx.x * blockDim.x + threadIdx.x) >> 5;
  int lane = threadIdx.x & 31;
  int m0 = wave * 16;
  if (m0 >= num_pad) return;
  int r = lane & 15, hh = lane >> 4;
  bool full = (m0 + 16 <= num);

  v16h ag0 = load_a_ws(aggh, m0, 0, lane);
  v16h ag1 = load_a_ws(aggh, m0, 32, lane);
  v16h hr0 = load_a_hr(h, Rbuf, m0, 0, lane, num);
  v16h hr1 = load_a_hr(h, Rbuf, m0, 32, lane, num);

#pragma unroll
  for (int nt = 0; nt < 4; ++nt) {
    int nc = nt * 16;
    int col = nc + r;
    v8f c = {};
    c = wmma_f16(ag0, load_b_lds<128>(swh, 0,  nc, lane), c);
    c = wmma_f16(ag1, load_b_lds<128>(swh, 32, nc, lane), c);
    c = wmma_f16(hr0, load_b_lds<128>(swh, 64, nc, lane), c);
    c = wmma_f16(hr1, load_b_lds<128>(swh, 96, nc, lane), c);
    float bias = blh[col];
    if (full) {
#pragma unroll
      for (int j = 0; j < 8; ++j) {
        size_t idx = (size_t)(m0 + hh * 8 + j) * EMBED + col;
        float Ht = tanhf(c[j] + bias);
        float z  = (float)Zbuf[idx];
        h0out[idx] = z * h[idx] + (1.0f - z) * Ht;
      }
    } else {
#pragma unroll
      for (int j = 0; j < 8; ++j) {
        int row = m0 + hh * 8 + j;
        if (row < num) {
          size_t idx = (size_t)row * EMBED + col;
          float Ht = tanhf(c[j] + bias);
          float z  = (float)Zbuf[idx];
          h0out[idx] = z * h[idx] + (1.0f - z) * Ht;
        }
      }
    }
  }
}

// ---------------------------------------------------------------------------
// K8: head = relu(h0) @ W_head + b_head  (wave per row, shuffle reduction)
// ---------------------------------------------------------------------------
__global__ void __launch_bounds__(256)
k_head(const float* __restrict__ h0, const float* __restrict__ Wh,
       const float* __restrict__ bh, float* __restrict__ out, int num) {
  int row  = (blockIdx.x * blockDim.x + threadIdx.x) >> 5;
  int lane = threadIdx.x & 31;
  if (row >= num) return;
  const float* p = h0 + (size_t)row * EMBED;
  float s = fmaxf(p[lane], 0.0f) * Wh[lane] +
            fmaxf(p[lane + 32], 0.0f) * Wh[lane + 32];
#pragma unroll
  for (int off = 16; off > 0; off >>= 1) s += __shfl_down(s, off, 32);
  if (lane == 0) out[row] = s + bh[0];
}

// ---------------------------------------------------------------------------
extern "C" void kernel_launch(void* const* d_in, const int* in_sizes, int n_in,
                              void* d_out, int out_size, void* d_ws, size_t ws_size,
                              hipStream_t stream) {
  const float* x    = (const float*)d_in[0];
  const int*   src  = (const int*)d_in[1];
  const int*   dst  = (const int*)d_in[2];
  const float* ew   = (const float*)d_in[3];
  const float* h    = (const float*)d_in[4];
  const float* Wcz  = (const float*)d_in[5];
  const float* bcz  = (const float*)d_in[6];
  const float* Wcr  = (const float*)d_in[7];
  const float* bcr  = (const float*)d_in[8];
  const float* Wch  = (const float*)d_in[9];
  const float* bch  = (const float*)d_in[10];
  const float* Wlz  = (const float*)d_in[11];
  const float* blz  = (const float*)d_in[12];
  const float* Wlr  = (const float*)d_in[13];
  const float* blr  = (const float*)d_in[14];
  const float* Wlh  = (const float*)d_in[15];
  const float* blh  = (const float*)d_in[16];
  const float* Whd  = (const float*)d_in[17];
  const float* bhd  = (const float*)d_in[18];

  int num     = in_sizes[0] / EMBED;     // B*N = 100000
  int num_pad = (num + 15) & ~15;        // padded row count for workspace
  int E       = in_sizes[1];             // 1.6M

  float* out      = (float*)d_out;
  float* head_out = out;                 // [num]
  float* h0_out   = out + num;           // [num, 64]

  // workspace (all padded to num_pad rows):
  //   dinv | agg{z,r,h} f32 | xw{z,r,h} f16 (reused for Z,R after scatter)
  char* w = (char*)d_ws;
  float* deg  = (float*)w;  w += (size_t)num_pad * sizeof(float);
  float* aggz = (float*)w;  w += (size_t)num_pad * EMBED * sizeof(float);
  float* aggr = (float*)w;  w += (size_t)num_pad * EMBED * sizeof(float);
  float* aggh = (float*)w;  w += (size_t)num_pad * EMBED * sizeof(float);
  _Float16* xwz = (_Float16*)w;  w += (size_t)num_pad * EMBED * sizeof(_Float16);
  _Float16* xwr = (_Float16*)w;  w += (size_t)num_pad * EMBED * sizeof(_Float16);
  _Float16* xwh = (_Float16*)w;  w += (size_t)num_pad * EMBED * sizeof(_Float16);
  _Float16* Zbuf = xwz;
  _Float16* Rbuf = xwr;

  int t256 = 256;
  k_deg_init<<<(num_pad + t256 - 1) / t256, t256, 0, stream>>>(deg, num_pad);
  k_deg_accum<<<(E + t256 - 1) / t256, t256, 0, stream>>>(deg, dst, ew, E);
  k_dinv<<<(num_pad + t256 - 1) / t256, t256, 0, stream>>>(deg, num_pad);

  int waves = num_pad / 16;              // one wave per 16 rows
  int gemm_blocks = (waves * 32 + 127) / 128;
  k_conv_gemm<<<gemm_blocks, 128, 0, stream>>>(
      x, deg, Wcz, bcz, Wcr, bcr, Wch, bch,
      xwz, xwr, xwh, aggz, aggr, aggh, num, num_pad);

  long long sc_threads = (long long)E * 32;
  k_edge_scatter<<<(int)((sc_threads + 255) / 256), 256, 0, stream>>>(
      src, dst, ew, deg, xwz, xwr, xwh, aggz, aggr, aggh, E);

  k_gate_zr<<<gemm_blocks, 128, 0, stream>>>(
      aggz, aggr, h, Wlz, blz, Wlr, blr, Zbuf, Rbuf, num, num_pad);

  k_gate_h<<<gemm_blocks, 128, 0, stream>>>(
      aggh, h, Zbuf, Rbuf, Wlh, blh, h0_out, num, num_pad);

  long long hd_threads = (long long)num * 32;
  k_head<<<(int)((hd_threads + 255) / 256), 256, 0, stream>>>(
      h0_out, Whd, bhd, head_out, num);
}
